// Attention_Layer_5196910428547
// MI455X (gfx1250) — matrix operand order repaired and checked by reading, not compiled, not run on hardware
//
#include <hip/hip_runtime.h>
#include <hip/hip_bf16.h>

#define S_LEN 4096
#define DIN   512
#define DOUT  256
#define NB    8

typedef __attribute__((ext_vector_type(16))) __bf16 v16bf;
typedef __attribute__((ext_vector_type(8)))  __bf16 v8bf;
typedef __attribute__((ext_vector_type(4)))  __bf16 v4bf;
typedef __attribute__((ext_vector_type(8)))  float  v8f;

static __device__ __forceinline__ v8f wmma_bf16(v16bf a, v16bf b, v8f c) {
  return __builtin_amdgcn_wmma_f32_16x16x32_bf16(
      false, a, false, b, (short)0, c, false, false);
}

static __device__ __forceinline__ v16bf cat8(v8bf lo, v8bf hi) {
  return __builtin_shufflevector(lo, hi, 0,1,2,3,4,5,6,7,8,9,10,11,12,13,14,15);
}

static __device__ __forceinline__ v8f vzero8() {
  v8f r;
#pragma unroll
  for (int i = 0; i < 8; ++i) r[i] = 0.f;
  return r;
}

// Async copy 16B/lane from global to LDS (gfx1250, ASYNCcnt-tracked).
static __device__ __forceinline__ void async_cp16(unsigned ldsaddr, const void* gaddr) {
  asm volatile("global_load_async_to_lds_b128 %0, %1, off"
               :: "v"(ldsaddr), "v"(gaddr) : "memory");
}

// ---------------------------------------------------------------------------
// Kernel 0: fp32 -> bf16 elementwise conversion (x and weights, done once).
// ---------------------------------------------------------------------------
__global__ __launch_bounds__(256) void cvt_bf16_kernel(
    const float* __restrict__ in, __bf16* __restrict__ outp, int n4)
{
  const int i = blockIdx.x * 256 + threadIdx.x;
  if (i < n4) {
    const float4 f = ((const float4*)in)[i];
    v4bf r;
    r[0] = (__bf16)f.x; r[1] = (__bf16)f.y; r[2] = (__bf16)f.z; r[3] = (__bf16)f.w;
    ((v4bf*)outp)[i] = r;
  }
}

// ---------------------------------------------------------------------------
// Kernel 1: QKV projection from bf16 operands. One wave -> one 16x16 tile of
// Q (biased, pre-scaled by 1/16), K (transposed to [B][D][S]), V.
// ---------------------------------------------------------------------------
__global__ __launch_bounds__(256) void qkv_proj_kernel(
    const __bf16* __restrict__ xb,
    const __bf16* __restrict__ Wqb, const __bf16* __restrict__ Wkb,
    const __bf16* __restrict__ Wvb,
    const float* __restrict__ bq, const float* __restrict__ bk,
    const float* __restrict__ bv,
    __bf16* __restrict__ Qb, __bf16* __restrict__ Kt, __bf16* __restrict__ Vb)
{
  const int lane = threadIdx.x & 31;
  const int wave = threadIdx.x >> 5;
  const int wt   = blockIdx.x * 8 + wave;   // 0 .. 32767
  const int row0 = (wt >> 4) << 4;
  const int n0   = (wt & 15) << 4;
  const int half = lane >> 4;
  const int l16  = lane & 15;

  v8f aq = vzero8(), ak = vzero8(), av = vzero8();

#pragma unroll
  for (int kk = 0; kk < 16; ++kk) {
    const int k0 = kk * 32;
    // A-operand 16x32, direct bf16 loads in A layout
    const __bf16* xr = xb + (size_t)(row0 + l16) * DIN + k0 + half * 8;
    v16bf a = cat8(*(const v8bf*)xr, *(const v8bf*)(xr + 16));
    // B-operand 32x16: lane = N column l16, elements over K in table order
    const size_t woff = (size_t)(k0 + half * 8) * DOUT + n0 + l16;
    v16bf wq, wk, wv;
#pragma unroll
    for (int s = 0; s < 16; ++s) {
      const size_t o2 = woff + (size_t)((s & 7) + (s >> 3) * 16) * DOUT;
      wq[s] = Wqb[o2]; wk[s] = Wkb[o2]; wv[s] = Wvb[o2];
    }
    aq = wmma_bf16(a, wq, aq);
    ak = wmma_bf16(a, wk, ak);
    av = wmma_bf16(a, wv, av);
  }

  const int col = n0 + l16;
  const float bqv = bq[col], bkv = bk[col], bvv = bv[col];
  const float scale = 0.0625f;  // 1/sqrt(256)
#pragma unroll
  for (int g = 0; g < 8; ++g) {
    const int grow  = row0 + g + half * 8;
    const int batch = grow >> 12;
    const int s     = grow & (S_LEN - 1);
    Qb[(size_t)grow * DOUT + col]                = (__bf16)((aq[g] + bqv) * scale);
    Kt[((size_t)batch * DOUT + col) * S_LEN + s] = (__bf16)(ak[g] + bkv);
    Vb[(size_t)grow * DOUT + col]                = (__bf16)(av[g] + bvv);
  }
}

// ---------------------------------------------------------------------------
// Kernel 2: flash attention. 8 waves/block share double-buffered K/V tiles in
// LDS (filled with global_load_async_to_lds_b128); each wave owns one
// 16-query tile with online softmax over 32 keys/iteration.
// LDS map (bytes): buf b at b*40960: K tile 256 rows x 80B, V tile at +20480,
//                  32 rows x 528B. P staging at 81920 + wave*1024 (16x32 bf16).
// ---------------------------------------------------------------------------
#define KROWB 80u
#define VROWB 528u
#define VBASE 20480u
#define BUFB  40960u
#define PBASE 81920u

__global__ __launch_bounds__(256) void flash_attn_kernel(
    const __bf16* __restrict__ Qb, const __bf16* __restrict__ Kt,
    const __bf16* __restrict__ Vb, float* __restrict__ out)
{
  __shared__ __align__(128) unsigned char smem[90112];
  const int lane = threadIdx.x & 31;
  const int wave = threadIdx.x >> 5;
  const int wt   = blockIdx.x * 8 + wave;   // 0 .. 2047
  const int batch = wt >> 8;
  const int s0    = (wt & 255) << 4;
  const int half  = lane >> 4;
  const int l16   = lane & 15;

  const __bf16* q  = Qb + ((size_t)batch * S_LEN + s0) * DOUT;
  const __bf16* kt = Kt + (size_t)batch * DOUT * S_LEN;
  const __bf16* v  = Vb + (size_t)batch * S_LEN * DOUT;
  const unsigned sbase = (unsigned)(uintptr_t)(&smem[0]);

  // Per-lane async-fill addressing (invariant part); j0 added per iteration.
  // K tile: 1024 transfers of 16B: t -> row d = t>>2, seg = t&3
  // V tile: 1024 transfers of 16B: t -> row key = t>>5, seg = t&31
  unsigned kldsoff[4], vldsoff[4];
  const __bf16 *kg[4], *vg[4];
#pragma unroll
  for (int qq = 0; qq < 4; ++qq) {
    const int t  = (wave * 4 + qq) * 32 + lane;
    const int kr = t >> 2, ks = t & 3;
    kldsoff[qq] = (unsigned)kr * KROWB + (unsigned)ks * 16u;
    kg[qq] = kt + (size_t)kr * S_LEN + ks * 8;
    const int vr = t >> 5, vs = t & 31;
    vldsoff[qq] = VBASE + (unsigned)vr * VROWB + (unsigned)vs * 16u;
    vg[qq] = v + (size_t)vr * DOUT + vs * 8;
  }

  // Q tile resident in registers, A-layout, 8 chunks of 16x32
  v16bf qa[8];
#pragma unroll
  for (int c = 0; c < 8; ++c) {
    const __bf16* qr = q + (size_t)l16 * DOUT + c * 32 + half * 8;
    qa[c] = cat8(*(const v8bf*)qr, *(const v8bf*)(qr + 16));
  }

  v8f o[16];
#pragma unroll
  for (int n = 0; n < 16; ++n) o[n] = vzero8();
  float m[8], l[8];
#pragma unroll
  for (int g = 0; g < 8; ++g) { m[g] = -1e30f; l[g] = 0.f; }

  // Prologue: stage tile 0 into buffer 0
#pragma unroll
  for (int qq = 0; qq < 4; ++qq) async_cp16(sbase + kldsoff[qq], kg[qq]);
#pragma unroll
  for (int qq = 0; qq < 4; ++qq) async_cp16(sbase + vldsoff[qq], vg[qq]);

  const int NIT = S_LEN / 32;
  for (int it = 0; it < NIT; ++it) {
    const int j0 = it * 32;
    const unsigned bb = (unsigned)(it & 1) * BUFB;

    if (it + 1 < NIT) {
      // stage next tile into the other buffer (its readers finished last iter)
      const unsigned nb = (unsigned)((it + 1) & 1) * BUFB;
      const int nj = j0 + 32;
#pragma unroll
      for (int qq = 0; qq < 4; ++qq)
        async_cp16(sbase + nb + kldsoff[qq], kg[qq] + nj);
#pragma unroll
      for (int qq = 0; qq < 4; ++qq)
        async_cp16(sbase + nb + vldsoff[qq], vg[qq] + (size_t)nj * DOUT);
      asm volatile("s_wait_asynccnt 0x8" ::: "memory");  // tile `it` done
    } else {
      asm volatile("s_wait_asynccnt 0x0" ::: "memory");
    }
    __syncthreads();  // all waves' fills of tile `it` complete

    // scores: two 16-key column tiles from LDS K tile (d-major, 80B rows)
    v8f sc0 = vzero8(), sc1 = vzero8();
#pragma unroll
    for (int c = 0; c < 8; ++c) {
      const unsigned ro = bb + (unsigned)(c * 32 + half * 8) * KROWB + (unsigned)l16 * 2u;
      v16bf b0, b1;
#pragma unroll
      for (int s = 0; s < 16; ++s) {
        const unsigned o2 = ro + (unsigned)((s & 7) + (s >> 3) * 16) * KROWB;
        b0[s] = *(const __bf16*)&smem[o2];
        b1[s] = *(const __bf16*)&smem[o2 + 32];
      }
      sc0 = wmma_bf16(qa[c], b0, sc0);
      sc1 = wmma_bf16(qa[c], b1, sc1);
    }

    // online softmax (row M = g + 8*half lives across the 16-lane half-wave)
    const unsigned pb = PBASE + (unsigned)wave * 1024u;
    float alpha[8];
#pragma unroll
    for (int g = 0; g < 8; ++g) {
      float xm = fmaxf(sc0[g], sc1[g]);
#pragma unroll
      for (int d = 1; d < 16; d <<= 1) xm = fmaxf(xm, __shfl_xor(xm, d, 32));
      const float mn = fmaxf(m[g], xm);
      const float a  = __expf(m[g] - mn);
      m[g] = mn; alpha[g] = a;
      const float p0 = __expf(sc0[g] - mn);
      const float p1 = __expf(sc1[g] - mn);
      const unsigned row = (unsigned)(g + half * 8);
      *(__bf16*)&smem[pb + row * 64u + (unsigned)l16 * 2u]       = (__bf16)p0;
      *(__bf16*)&smem[pb + row * 64u + 32u + (unsigned)l16 * 2u] = (__bf16)p1;
      float ps = p0 + p1;
#pragma unroll
      for (int d = 1; d < 16; d <<= 1) ps += __shfl_xor(ps, d, 32);
      l[g] = l[g] * a + ps;
    }

#pragma unroll
    for (int n = 0; n < 16; ++n)
#pragma unroll
      for (int g = 0; g < 8; ++g) o[n][g] *= alpha[g];

    // P back in A-layout (same-wave LDS is in-order)
    asm volatile("s_wait_dscnt 0x0" ::: "memory");
    const unsigned pr = pb + (unsigned)l16 * 64u + (unsigned)half * 16u;
    v16bf pa = cat8(*(const v8bf*)&smem[pr], *(const v8bf*)&smem[pr + 32]);

    // O += P(16x32) * V(32x16) per 16-d chunk, from LDS V tile (528B rows)
    const unsigned vb0 = bb + VBASE + (unsigned)(half * 8) * VROWB + (unsigned)l16 * 2u;
#pragma unroll
    for (int n = 0; n < 16; ++n) {
      const unsigned vo = vb0 + (unsigned)n * 32u;
      v16bf bv;
#pragma unroll
      for (int s = 0; s < 16; ++s)
        bv[s] = *(const __bf16*)&smem[vo + (unsigned)((s & 7) + (s >> 3) * 16) * VROWB];
      o[n] = wmma_bf16(pa, bv, o[n]);
    }
    __syncthreads();  // reads of tile `it` done before its buffer is refilled
  }

  float inv[8];
#pragma unroll
  for (int g = 0; g < 8; ++g) inv[g] = 1.f / l[g];
  float* op = out + ((size_t)batch * S_LEN + s0) * DOUT;
#pragma unroll
  for (int n = 0; n < 16; ++n)
#pragma unroll
    for (int g = 0; g < 8; ++g)
      op[(size_t)(g + half * 8) * DOUT + n * 16 + l16] = o[n][g] * inv[g];
}

extern "C" void kernel_launch(void* const* d_in, const int* in_sizes, int n_in,
                              void* d_out, int out_size, void* d_ws, size_t ws_size,
                              hipStream_t stream) {
  const float* x  = (const float*)d_in[0];
  const float* Wq = (const float*)d_in[1];
  const float* bq = (const float*)d_in[2];
  const float* Wk = (const float*)d_in[3];
  const float* bk = (const float*)d_in[4];
  const float* Wv = (const float*)d_in[5];
  const float* bv = (const float*)d_in[6];
  float* out = (float*)d_out;

  const size_t NQ = (size_t)NB * S_LEN * DOUT;   // 8.4M elements
  const size_t XN = (size_t)NB * S_LEN * DIN;    // 16.8M
  const size_t WN = (size_t)DIN * DOUT;          // 131K

  __bf16* Qb  = (__bf16*)d_ws;
  __bf16* Kt  = Qb + NQ;
  __bf16* Vb  = Kt + NQ;
  __bf16* xb  = Vb + NQ;
  __bf16* Wqb = xb + XN;
  __bf16* Wkb = Wqb + WN;
  __bf16* Wvb = Wkb + WN;

  cvt_bf16_kernel<<<(int)(XN / 4 / 256), 256, 0, stream>>>(x, xb, (int)(XN / 4));
  cvt_bf16_kernel<<<(int)(WN / 4 / 256), 256, 0, stream>>>(Wq, Wqb, (int)(WN / 4));
  cvt_bf16_kernel<<<(int)(WN / 4 / 256), 256, 0, stream>>>(Wk, Wkb, (int)(WN / 4));
  cvt_bf16_kernel<<<(int)(WN / 4 / 256), 256, 0, stream>>>(Wv, Wvb, (int)(WN / 4));

  qkv_proj_kernel<<<4096, 256, 0, stream>>>(xb, Wqb, Wkb, Wvb, bq, bk, bv, Qb, Kt, Vb);
  flash_attn_kernel<<<256, 256, 0, stream>>>(Qb, Kt, Vb, out);
}
